// RelationModel_48189533061762
// MI455X (gfx1250) — compile-verified, matrix operand order checked by reading, not run on hardware
//
#include <hip/hip_runtime.h>
#include <cstdint>

#define Bq 32
#define Lq 8
#define Nq 128
#define Cq 1024
#define Eq 64
#define Aq 64
#define Hq 256
#define Dq 1152   // C + N

typedef float   v8f   __attribute__((ext_vector_type(8)));
typedef __bf16  v16bf __attribute__((ext_vector_type(16)));
typedef unsigned short v16us __attribute__((ext_vector_type(16)));
typedef unsigned int v4ui __attribute__((ext_vector_type(4)));
typedef int v8i __attribute__((ext_vector_type(8)));
typedef int v4i __attribute__((ext_vector_type(4)));

union Frag {
    v16bf bf;
    v16us us;
    unsigned u32[8];
};

__device__ __forceinline__ unsigned short f2bf(float f) {
    unsigned u = __float_as_uint(f);
    unsigned r = u + 0x7FFFu + ((u >> 16) & 1u);   // round-to-nearest-even
    return (unsigned short)(r >> 16);
}

__device__ __forceinline__ void ldfrag(Frag& f, const unsigned short* p0, const unsigned short* p1) {
    uint4 x = *(const uint4*)p0;
    uint4 y = *(const uint4*)p1;
    f.u32[0] = x.x; f.u32[1] = x.y; f.u32[2] = x.z; f.u32[3] = x.w;
    f.u32[4] = y.x; f.u32[5] = y.y; f.u32[6] = y.z; f.u32[7] = y.w;
}

__device__ __forceinline__ v8f wmma_bf16(const Frag& a, const Frag& b, v8f c) {
    return __builtin_amdgcn_wmma_f32_16x16x32_bf16(false, a.bf, false, b.bf, (short)0, c, false, false);
}

// Per-wave LDS store->load ordering: DS ops from one wave execute in order
// (ISA 7.3), we only need to stop compiler reordering across the round trip.
__device__ __forceinline__ void wave_lds_fence() {
    asm volatile("" ::: "memory");
    __builtin_amdgcn_wave_barrier();
    asm volatile("" ::: "memory");
}

// TDM: load `rows` x 64 bf16 tile (row stride 64 elements) global -> LDS.
__device__ __forceinline__ void tdm_load_tile(unsigned lds_addr, const unsigned short* gaddr, int rows) {
    unsigned long long ga = (unsigned long long)(uintptr_t)gaddr;
    v4ui g0;
    g0.x = 1u;                                                  // count = 1 (valid)
    g0.y = lds_addr;                                            // lds_addr
    g0.z = (unsigned)ga;                                        // global_addr lo
    g0.w = (unsigned)((ga >> 32) & 0x01FFFFFFu) | 0x80000000u;  // global_addr hi | type=2
    v8i g1;
    g1[0] = 0x00010000;                   // data_size = 2 bytes
    g1[1] = (int)(64u << 16);             // tensor_dim0 = 64
    g1[2] = (int)((unsigned)rows << 16);  // tensor_dim1 = rows
    g1[3] = (int)(64u << 16);             // tile_dim0 = 64
    g1[4] = rows;                         // tile_dim1 = rows, tile_dim2 = 0
    g1[5] = 64;                           // tensor_dim0_stride = 64
    g1[6] = 0;
    g1[7] = 0;
    v4i gz = {0, 0, 0, 0};
#if __clang_major__ >= 23
    v8i gz8 = {0, 0, 0, 0, 0, 0, 0, 0};
    __builtin_amdgcn_tensor_load_to_lds(g0, g1, gz, gz, gz8, 0);
#else
    __builtin_amdgcn_tensor_load_to_lds(g0, g1, gz, gz, 0);
#endif
}

// ---------------------------------------------------------------------------
// Build dendron (bf16, into da cols 0:64), axon (f32 + bf16), attentions[0].
// ---------------------------------------------------------------------------
__global__ void __launch_bounds__(256) k_init(
    const int* __restrict__ gt_classes, const int* __restrict__ gt_attributes,
    const float* __restrict__ cin, const float* __restrict__ ain, const float* __restrict__ uin,
    const float* __restrict__ cout, const float* __restrict__ aout, const float* __restrict__ uout,
    const float* __restrict__ att_init,
    float* __restrict__ axon, unsigned short* __restrict__ axbf,
    unsigned short* __restrict__ da, float* __restrict__ att0)
{
    int rr = blockIdx.x * 4 + (threadIdx.x >> 6);
    int e  = threadIdx.x & 63;
    int b  = rr / Dq, d = rr % Dq;
    float vin, vout;
    if (d < Cq) {
        vin  = uin[d * Eq + e];
        vout = uout[d * Eq + e];
    } else {
        int n = d - Cq;
        int c = gt_classes[b * Nq + n] + 1;
        vin  = cin[c * Eq + e];
        vout = cout[c * Eq + e];
        #pragma unroll
        for (int k = 0; k < 8; ++k) {
            int a = gt_attributes[(b * Nq + n) * 8 + k];
            if (a != -1) {
                vin  += ain[(a + 1) * Eq + e];
                vout += aout[(a + 1) * Eq + e];
            }
        }
    }
    size_t r = (size_t)rr;
    axon[r * Eq + e]  = vout;
    axbf[r * Eq + e]  = f2bf(vout);
    da[r * 128 + e]   = f2bf(vin);
    att0[r * Aq + e]  = att_init[e];
}

// ---------------------------------------------------------------------------
// One-time small init: meta, arguments, transposed bf16 axon-MLP weights.
// ---------------------------------------------------------------------------
__global__ void __launch_bounds__(256) k_small(
    const float* __restrict__ meta_init, const int* __restrict__ program_arg,
    const float* __restrict__ uout, const float* __restrict__ w1, const float* __restrict__ w2,
    float* __restrict__ meta, float* __restrict__ args,
    unsigned short* __restrict__ w1t, unsigned short* __restrict__ w2t)
{
    int t = threadIdx.x;
    for (int i = t; i < Bq * Aq; i += 256) meta[i] = meta_init[i & 63];
    for (int i = t; i < Bq * Lq * Eq; i += 256) {
        int e = i & 63, bl = i >> 6;
        args[i] = uout[program_arg[bl] * Eq + e];
    }
    for (int i = t; i < Hq * 128; i += 256) {      // w1t[h][j] = w1[j][h]
        int h = i >> 7, j = i & 127;
        w1t[i] = f2bf(w1[j * Hq + h]);
    }
    for (int i = t; i < Aq * Hq; i += 256) {       // w2t[a][h] = w2[h][a]
        int a = i >> 8, h = i & 255;
        w2t[i] = f2bf(w2[h * Aq + a]);
    }
}

// ---------------------------------------------------------------------------
// Per-iteration: att_mean, ax_mean, attention->bf16 into da cols 64:128.
// ---------------------------------------------------------------------------
__global__ void __launch_bounds__(256) k_pre(
    const float* __restrict__ att, const float* __restrict__ axon,
    const float* __restrict__ args,
    float* __restrict__ att_mean, float* __restrict__ ax_mean,
    unsigned short* __restrict__ da, int iter)
{
    int r    = blockIdx.x * 8 + (threadIdx.x >> 5);
    int lane = threadIdx.x & 31;
    int b    = r / Dq;
    const float* ap = att + (size_t)r * Aq;
    float a0 = ap[lane], a1 = ap[lane + 32];
    da[(size_t)r * 128 + 64 + lane] = f2bf(a0);
    da[(size_t)r * 128 + 96 + lane] = f2bf(a1);
    float s = a0 + a1;
    const float* xp = axon + (size_t)r * Eq;
    const float* gp = args + (size_t)(b * Lq + iter) * Eq;
    float t2 = xp[lane] * gp[lane] + xp[lane + 32] * gp[lane + 32];
    #pragma unroll
    for (int off = 16; off; off >>= 1) {
        s  += __shfl_down(s, off, 32);
        t2 += __shfl_down(t2, off, 32);
    }
    if (lane == 0) {
        att_mean[r] = s * (1.0f / 64.0f);
        ax_mean[r]  = t2 * (1.0f / 64.0f);
    }
}

// ---------------------------------------------------------------------------
// Meta MLP (32 rows, 128->256->64), conditional update on OP_MODE.
// ---------------------------------------------------------------------------
__global__ void __launch_bounds__(256) k_meta(
    float* __restrict__ meta, const float* __restrict__ args,
    const int* __restrict__ program_op,
    const float* __restrict__ w1, const float* __restrict__ bb1,
    const float* __restrict__ w2, const float* __restrict__ bb2, int iter)
{
    __shared__ float x[128];
    __shared__ float h[256];
    int b = blockIdx.x, t = threadIdx.x;
    if (t < 64)       x[t] = meta[b * Aq + t];
    else if (t < 128) x[t] = args[(b * Lq + iter) * Eq + (t - 64)];
    __syncthreads();
    float acc = bb1[t];
    for (int j = 0; j < 128; ++j) acc += x[j] * w1[j * Hq + t];
    h[t] = fmaxf(acc, 0.0f);
    __syncthreads();
    if (t < 64) {
        float o = bb2[t];
        for (int j = 0; j < 256; ++j) o += h[j] * w2[j * Aq + t];
        float m = (program_op[b * Lq + iter] == 0) ? 1.0f : 0.0f;
        meta[b * Aq + t] = m * o + (1.0f - m) * x[t];
    }
}

// ---------------------------------------------------------------------------
// Main fused kernel. 8 waves x 16 rows of D per block; double-buffered k-loop:
//   tile ks+1 staged (TDM + async-LDS + reg prefetch) while computing tile ks.
// ---------------------------------------------------------------------------
__global__ void __launch_bounds__(256) k_main(
    const unsigned short* __restrict__ da,    // (B,D,128) bf16 [dendron|attention]
    const unsigned short* __restrict__ axbf,  // (B,D,64)  bf16 axon
    const float* __restrict__ att_mean,       // (B,D)
    const float* __restrict__ ax_mean,        // (B,D)
    const float* __restrict__ meta,           // (B,A)
    const float* __restrict__ att_prev,       // (B,D,A)
    float* __restrict__ att_next,             // (B,D,A)
    const unsigned short* __restrict__ w1t,   // (256,128) bf16
    const unsigned short* __restrict__ w2t,   // (64,256)  bf16
    const float* __restrict__ b1, const float* __restrict__ b2,
    const int* __restrict__ program_op, int iter)
{
    const int b    = blockIdx.y;
    const int tid  = threadIdx.x;
    const int wave = tid >> 5;
    const int lane = tid & 31;
    const int lr   = lane & 15;
    const int hi   = lane >> 4;
    const int cb   = hi * 8;
    const int m0   = blockIdx.x * 128 + wave * 16;

    __shared__ __align__(16) unsigned short aaS[2][32 * 64];     // axon k-tile (row major)
    __shared__ __align__(16) unsigned short datrS[2][128 * 32];  // d_a k-tile transposed [f][k]
    __shared__ __align__(16) float wscrS[8][16 * 32];            // per-wave transpose scratch
    float* wscr = wscrS[wave];

    // A fragments: dendron rows m0..m0+15 (bf16, cols 0:64 of da)
    const unsigned short* drow = da + ((size_t)(b * Dq + m0 + lr)) * 128;
    Frag a0, a1;
    ldfrag(a0, drow + cb,      drow + cb + 16);
    ldfrag(a1, drow + 32 + cb, drow + 48 + cb);

    v8f zero = {0.f, 0.f, 0.f, 0.f, 0.f, 0.f, 0.f, 0.f};
    v8f macc[8];
    #pragma unroll
    for (int t = 0; t < 8; ++t) macc[t] = zero;

    const float invD = 1.0f / (float)Dq;
    const int kkp = tid & 15;        // k-row pair for datr staging
    const int fg  = tid >> 4;        // f-group (8 cols)

    // ---- staging helpers ---------------------------------------------------
    auto issue_aa = [&](int ks, int buf) {
        const unsigned short* gbase = axbf + ((size_t)(b * Dq + ks * 32)) * Eq;
        if (wave == 0) {
            // rows 0..15 via Tensor Data Mover
            tdm_load_tile((unsigned)(uintptr_t)&aaS[buf][0], gbase, 16);
        } else if (wave >= 4) {
            // rows 16..31 via async global->LDS (128 lanes x 16B = 2KB)
            int i = tid - 128;
            unsigned ldsoff = (unsigned)(uintptr_t)&aaS[buf][0] + 2048u + (unsigned)i * 16u;
            unsigned goff   = 2048u + (unsigned)i * 16u;
            asm volatile("global_load_async_to_lds_b128 %0, %1, %2 offset:0"
                         :: "v"(ldsoff), "v"(goff), "s"(gbase) : "memory");
        }
    };
    auto preload_datr = [&](int ks, uint4& r0, uint4& r1) {
        const unsigned short* p = da + ((size_t)(b * Dq + ks * 32 + 2 * kkp)) * 128 + fg * 8;
        r0 = *(const uint4*)p;
        r1 = *(const uint4*)(p + 128);
    };
    auto store_datr = [&](int buf, uint4 r0, uint4 r1) {
        unsigned* dst = (unsigned*)&datrS[buf][0];
        unsigned lo[4] = {r0.x, r0.y, r0.z, r0.w};
        unsigned hi4[4] = {r1.x, r1.y, r1.z, r1.w};
        #pragma unroll
        for (int j = 0; j < 8; ++j) {
            unsigned lov = (lo[j >> 1] >> ((j & 1) * 16)) & 0xFFFFu;
            unsigned hiv = (hi4[j >> 1] << (16 - (j & 1) * 16)) & 0xFFFF0000u;
            dst[(fg * 8 + j) * 16 + kkp] = lov | hiv;   // [f][k] packed pair
        }
    };

    // ---- prologue: stage tile 0 --------------------------------------------
    uint4 r0, r1;
    preload_datr(0, r0, r1);
    issue_aa(0, 0);

    for (int ks = 0; ks < Dq / 32; ++ks) {
        const int cur = ks & 1, nxt = cur ^ 1;
        store_datr(cur, r0, r1);
        if (wave == 0)      __builtin_amdgcn_s_wait_tensorcnt(0);
        else if (wave >= 4) asm volatile("s_wait_asynccnt 0" ::: "memory");
        __syncthreads();
        if (ks + 1 < Dq / 32) {               // prefetch next tile during compute
            preload_datr(ks + 1, r0, r1);
            issue_aa(ks + 1, nxt);
        }

        const unsigned short* aa  = &aaS[cur][0];
        const unsigned short* dtr = &datrS[cur][0];

        // GEMM1: c0t = dendron_tile . axon[k..k+15]^T ; c1t = ... [k+16..k+31]
        v8f c0t = zero, c1t = zero;
        {
            Frag bf_;
            ldfrag(bf_, &aa[lr * 64 + cb],             &aa[lr * 64 + cb + 16]);
            c0t = wmma_bf16(a0, bf_, c0t);
            ldfrag(bf_, &aa[lr * 64 + 32 + cb],        &aa[lr * 64 + 48 + cb]);
            c0t = wmma_bf16(a1, bf_, c0t);
            ldfrag(bf_, &aa[(16 + lr) * 64 + cb],      &aa[(16 + lr) * 64 + cb + 16]);
            c1t = wmma_bf16(a0, bf_, c1t);
            ldfrag(bf_, &aa[(16 + lr) * 64 + 32 + cb], &aa[(16 + lr) * 64 + 48 + cb]);
            c1t = wmma_bf16(a1, bf_, c1t);
        }
        // fold activated_axon scale (att_mean[k]) and 1/D into w; clip to [-1,1]
        float s0 = att_mean[b * Dq + ks * 32 + lr] * invD;
        float s1 = att_mean[b * Dq + ks * 32 + 16 + lr] * invD;
        #pragma unroll
        for (int r = 0; r < 8; ++r) {
            c0t[r] = fminf(fmaxf(c0t[r] * s0, -1.0f), 1.0f);
            c1t[r] = fminf(fmaxf(c1t[r] * s1, -1.0f), 1.0f);
        }
        // transpose w tile (C layout -> A layout) through per-wave LDS scratch
        #pragma unroll
        for (int r = 0; r < 8; ++r) {
            wscr[(r + 8 * hi) * 32 + lr]      = c0t[r];
            wscr[(r + 8 * hi) * 32 + 16 + lr] = c1t[r];
        }
        wave_lds_fence();
        Frag aw;
        #pragma unroll
        for (int j = 0; j < 8; ++j) {
            aw.us[j]     = f2bf(wscr[lr * 32 + cb + j]);
            aw.us[8 + j] = f2bf(wscr[lr * 32 + cb + 16 + j]);
        }
        wave_lds_fence();
        // GEMM2: message += w_tile . dendron_attention_k (8 column tiles of 16)
        #pragma unroll
        for (int ft = 0; ft < 8; ++ft) {
            Frag bf_;
            ldfrag(bf_, &dtr[(ft * 16 + lr) * 32 + cb], &dtr[(ft * 16 + lr) * 32 + cb + 16]);
            macc[ft] = wmma_bf16(aw, bf_, macc[ft]);
        }
    }

    #pragma unroll
    for (int t = 0; t < 8; ++t)
        #pragma unroll
        for (int r = 0; r < 8; ++r) macc[t][r] *= invD;

    // message -> A fragments (4 x 16x32) via scratch round trips
    Frag mf[4];
    #pragma unroll
    for (int t = 0; t < 4; ++t) {
        wave_lds_fence();
        #pragma unroll
        for (int r = 0; r < 8; ++r) {
            wscr[(r + 8 * hi) * 32 + lr]      = macc[2 * t][r];
            wscr[(r + 8 * hi) * 32 + 16 + lr] = macc[2 * t + 1][r];
        }
        wave_lds_fence();
        #pragma unroll
        for (int j = 0; j < 8; ++j) {
            mf[t].us[j]     = f2bf(wscr[lr * 32 + cb + j]);
            mf[t].us[8 + j] = f2bf(wscr[lr * 32 + cb + 16 + j]);
        }
    }

    // axon MLP: hidden chunks of 32, accumulate 16x64 output
    v8f oacc[4];
    #pragma unroll
    for (int t = 0; t < 4; ++t) oacc[t] = zero;

    for (int hc = 0; hc < 8; ++hc) {
        int h0 = hc * 32;
        v8f h0t = zero, h1t = zero;
        #pragma unroll
        for (int jc = 0; jc < 4; ++jc) {
            Frag bf_;
            const unsigned short* p0 = w1t + (size_t)(h0 + lr) * 128 + jc * 32 + cb;
            ldfrag(bf_, p0, p0 + 16);
            h0t = wmma_bf16(mf[jc], bf_, h0t);
            const unsigned short* p1 = w1t + (size_t)(h0 + 16 + lr) * 128 + jc * 32 + cb;
            ldfrag(bf_, p1, p1 + 16);
            h1t = wmma_bf16(mf[jc], bf_, h1t);
        }
        float bb0 = b1[h0 + lr], bb1v = b1[h0 + 16 + lr];
        #pragma unroll
        for (int r = 0; r < 8; ++r) {
            h0t[r] = fmaxf(h0t[r] + bb0, 0.0f);
            h1t[r] = fmaxf(h1t[r] + bb1v, 0.0f);
        }
        wave_lds_fence();
        #pragma unroll
        for (int r = 0; r < 8; ++r) {
            wscr[(r + 8 * hi) * 32 + lr]      = h0t[r];
            wscr[(r + 8 * hi) * 32 + 16 + lr] = h1t[r];
        }
        wave_lds_fence();
        Frag ah;
        #pragma unroll
        for (int j = 0; j < 8; ++j) {
            ah.us[j]     = f2bf(wscr[lr * 32 + cb + j]);
            ah.us[8 + j] = f2bf(wscr[lr * 32 + cb + 16 + j]);
        }
        #pragma unroll
        for (int ot = 0; ot < 4; ++ot) {
            Frag bf_;
            const unsigned short* p = w2t + (size_t)(ot * 16 + lr) * 256 + h0 + cb;
            ldfrag(bf_, p, p + 16);
            oacc[ot] = wmma_bf16(ah, bf_, oacc[ot]);
        }
    }

    // epilogue: attention = relu(prev + ii*insert + it*transfer)
    int op = program_op[b * Lq + iter];
    float fi  = (op == 1) ? 1.0f : 0.0f;
    float ftv = (op == 2) ? 1.0f : 0.0f;
    #pragma unroll
    for (int ot = 0; ot < 4; ++ot) {
        int a = ot * 16 + lr;
        float mta = meta[b * Aq + a];
        float bba = b2[a];
        #pragma unroll
        for (int r = 0; r < 8; ++r) {
            int m = m0 + r + 8 * hi;
            float transfer = oacc[ot][r] + bba + macc[4 + ot][r];
            float insert   = mta * ax_mean[b * Dq + m];
            float prev     = att_prev[((size_t)(b * Dq + m)) * Aq + a];
            att_next[((size_t)(b * Dq + m)) * Aq + a] =
                fmaxf(prev + fi * insert + ftv * transfer, 0.0f);
        }
    }
}

// ---------------------------------------------------------------------------
// out = log_softmax over D of attention.mean(axis=A), per batch.
// ---------------------------------------------------------------------------
__global__ void __launch_bounds__(256) k_final(const float* __restrict__ attL, float* __restrict__ out)
{
    __shared__ float v[Dq];
    __shared__ float red[8];
    int b = blockIdx.x, t = threadIdx.x;
    for (int d = t; d < Dq; d += 256) {
        const float* p = attL + ((size_t)(b * Dq + d)) * Aq;
        float s = 0.0f;
        for (int a = 0; a < Aq; ++a) s += p[a];
        v[d] = s * (1.0f / 64.0f);
    }
    __syncthreads();
    float mx = -1e30f;
    for (int d = t; d < Dq; d += 256) mx = fmaxf(mx, v[d]);
    #pragma unroll
    for (int off = 16; off; off >>= 1) mx = fmaxf(mx, __shfl_down(mx, off, 32));
    if ((t & 31) == 0) red[t >> 5] = mx;
    __syncthreads();
    if (t == 0) { float m2 = red[0]; for (int i = 1; i < 8; ++i) m2 = fmaxf(m2, red[i]); red[0] = m2; }
    __syncthreads();
    mx = red[0];
    float se = 0.0f;
    for (int d = t; d < Dq; d += 256) se += expf(v[d] - mx);
    #pragma unroll
    for (int off = 16; off; off >>= 1) se += __shfl_down(se, off, 32);
    __syncthreads();
    if ((t & 31) == 0) red[t >> 5] = se;
    __syncthreads();
    if (t == 0) { float s2 = 0.0f; for (int i = 0; i < 8; ++i) s2 += red[i]; red[0] = s2; }
    __syncthreads();
    float lse = mx + logf(red[0]);
    for (int d = t; d < Dq; d += 256) out[b * Dq + d] = v[d] - lse;
}

// ---------------------------------------------------------------------------
extern "C" void kernel_launch(void* const* d_in, const int* in_sizes, int n_in,
                              void* d_out, int out_size, void* d_ws, size_t ws_size,
                              hipStream_t stream)
{
    const int*   program_op    = (const int*)d_in[0];
    const int*   program_arg   = (const int*)d_in[1];
    const int*   gt_classes    = (const int*)d_in[2];
    const int*   gt_attributes = (const int*)d_in[3];
    const float* class_embIn   = (const float*)d_in[4];
    const float* attr_embIn    = (const float*)d_in[5];
    const float* concept_embIn = (const float*)d_in[6];
    const float* class_embOut  = (const float*)d_in[7];
    const float* attr_embOut   = (const float*)d_in[8];
    const float* concept_embOut= (const float*)d_in[9];
    const float* meta_init     = (const float*)d_in[10];
    const float* attention_init= (const float*)d_in[11];
    const float* axon_w1       = (const float*)d_in[12];
    const float* axon_b1       = (const float*)d_in[13];
    const float* axon_w2       = (const float*)d_in[14];
    const float* axon_b2       = (const float*)d_in[15];
    const float* meta_w1       = (const float*)d_in[16];
    const float* meta_b1       = (const float*)d_in[17];
    const float* meta_w2       = (const float*)d_in[18];
    const float* meta_b2       = (const float*)d_in[19];

    float* out  = (float*)d_out;
    float* atts = out + (size_t)Bq * Dq;   // (L+1, B, D, A)

    char* ws = (char*)d_ws;
    size_t off = 0;
    auto alloc = [&](size_t bytes) { void* p = ws + off; off = (off + bytes + 255) & ~(size_t)255; return p; };
    float*          axon     = (float*)alloc((size_t)Bq * Dq * Eq * 4);
    unsigned short* axbf     = (unsigned short*)alloc((size_t)Bq * Dq * Eq * 2);
    unsigned short* da       = (unsigned short*)alloc((size_t)Bq * Dq * 128 * 2);
    float*          att_mean = (float*)alloc((size_t)Bq * Dq * 4);
    float*          ax_mean  = (float*)alloc((size_t)Bq * Dq * 4);
    float*          meta     = (float*)alloc((size_t)Bq * Aq * 4);
    float*          args     = (float*)alloc((size_t)Bq * Lq * Eq * 4);
    unsigned short* w1t      = (unsigned short*)alloc((size_t)Hq * 128 * 2);
    unsigned short* w2t      = (unsigned short*)alloc((size_t)Aq * Hq * 2);

    k_small<<<1, 256, 0, stream>>>(meta_init, program_arg, concept_embOut,
                                   axon_w1, axon_w2, meta, args, w1t, w2t);
    k_init<<<Bq * Dq / 4, 256, 0, stream>>>(gt_classes, gt_attributes,
                                            class_embIn, attr_embIn, concept_embIn,
                                            class_embOut, attr_embOut, concept_embOut,
                                            attention_init, axon, axbf, da, atts);

    for (int i = 0; i < Lq; ++i) {
        const float* attp = atts + (size_t)i * Bq * Dq * Aq;
        float*       attn = atts + (size_t)(i + 1) * Bq * Dq * Aq;
        k_pre<<<Bq * Dq / 8, 256, 0, stream>>>(attp, axon, args, att_mean, ax_mean, da, i);
        k_meta<<<Bq, 256, 0, stream>>>(meta, args, program_op,
                                       meta_w1, meta_b1, meta_w2, meta_b2, i);
        k_main<<<dim3(Dq / 128, Bq), 256, 0, stream>>>(da, axbf, att_mean, ax_mean, meta,
                                                       attp, attn, w1t, w2t,
                                                       axon_b1, axon_b2, program_op, i);
    }
    k_final<<<Bq, 256, 0, stream>>>(atts + (size_t)Lq * Bq * Dq * Aq, out);
}
